// TtT5Attention_17257178595897
// MI455X (gfx1250) — matrix operand order repaired and checked by reading, not compiled, not run on hardware
//
#include <hip/hip_runtime.h>
#include <hip/hip_bf16.h>
#include <math.h>

typedef __bf16 bf16_t;
typedef __attribute__((ext_vector_type(8)))  __bf16 v8bf;
typedef __attribute__((ext_vector_type(16))) __bf16 v16bf;
typedef __attribute__((ext_vector_type(8)))  float  v8f;

#define B_  8
#define S_  1024
#define D_  1024
#define H_  16
#define DK_ 64

// ---------------------------------------------------------------------------
// WMMA helpers (CDNA5, wave32). D = A(16x32 bf16) x B(32x16 bf16) + C(16x16 f32)
// ---------------------------------------------------------------------------
static __device__ __forceinline__ v8f wmma_bf16(v16bf a, v16bf b, v8f c) {
  return __builtin_amdgcn_wmma_f32_16x16x32_bf16(
      /*neg_a=*/false, a, /*neg_b=*/false, b,
      /*c_mod=*/(short)0, c, /*reuse_a=*/false, /*reuse_b=*/false);
}

// A-operand fragment: lane holds row (lane&15); K contiguous {2*kb..2*kb+15},
// kb = 0 for lanes 0-15, 8 for lanes 16-31 (matches contiguous B order).
static __device__ __forceinline__ v16bf ld_a_frag(const bf16_t* p, int kbA) {
  v8bf lo = *(const v8bf*)(p + 2 * kbA);
  v8bf hi = *(const v8bf*)(p + 2 * kbA + 8);
  return __builtin_shufflevector(lo, hi, 0, 1, 2, 3, 4, 5, 6, 7,
                                         8, 9, 10, 11, 12, 13, 14, 15);
}

// ---------------------------------------------------------------------------
// Prep kernels
// ---------------------------------------------------------------------------
__global__ void __launch_bounds__(256)
k_cvt_bf16(const float* __restrict__ x, bf16_t* __restrict__ y, int n) {
  for (int i = blockIdx.x * blockDim.x + threadIdx.x; i < n;
       i += gridDim.x * blockDim.x)
    y[i] = (bf16_t)x[i];
}

// wt[n][k] = w[k][n]  (1024x1024), output-indexed for coalesced writes
__global__ void __launch_bounds__(256)
k_transpose_w(const float* __restrict__ w, bf16_t* __restrict__ wt) {
  int i = blockIdx.x * blockDim.x + threadIdx.x;
  if (i >= 1024 * 1024) return;
  int n = i >> 10, k = i & 1023;
  wt[i] = (bf16_t)w[k * 1024 + n];
}

// T5 relative position bias table: tab[h*2047 + (delta+1023)], delta = mem-ctx
__global__ void __launch_bounds__(256)
k_bias_tab(const float* __restrict__ rel_bias, float* __restrict__ tab) {
  int i = blockIdx.x * blockDim.x + threadIdx.x;
  if (i >= H_ * 2047) return;
  int h = i / 2047, d = i % 2047;
  int delta = d - 1023;
  int rb = (delta > 0) ? 16 : 0;           // num_buckets//2 = 16 (bidirectional)
  int arp = (delta < 0) ? -delta : delta;
  int bucket;
  if (arp < 8) {                           // max_exact = 8
    bucket = rb + arp;
  } else {
    // 8 + log(arp/8)/log(128/8) * 8, clamped to 15
    int lg = 8 + (int)(logf((float)arp * 0.125f) * (8.0f / logf(16.0f)));
    bucket = rb + (lg > 15 ? 15 : lg);
  }
  tab[i] = rel_bias[bucket * H_ + h];
}

// ---------------------------------------------------------------------------
// Generic bf16 WMMA GEMM:  C[M,N] = A[M,K] * Bt[N,K]^T
// Block tile 128x128, 8 waves (4 along M x 2 along N), wave tile 32x64.
// mode 0: out = f32 row-major [M,N]       (final projection)
// mode 1: out = bf16 [b,h,s,dk]           (Q / K operand layout)
// mode 2: out = bf16 [b,h,dk,s]           (V^T operand layout)
// ---------------------------------------------------------------------------
__global__ void __launch_bounds__(256)
k_gemm_bf16(const bf16_t* __restrict__ A, const bf16_t* __restrict__ Bt,
            void* __restrict__ out, int M, int N, int K, int mode) {
  const int lane  = threadIdx.x & 31;
  const int wave  = threadIdx.x >> 5;
  const int lr    = lane & 15;
  const int kbA   = (lane < 16) ? 0 : 8;    // A half-lane K offset
  const int kbB   = (lane < 16) ? 0 : 16;   // B half-lane K offset (contiguous 16)
  const int rowoff = (lane < 16) ? 0 : 8;   // C/D layout: lanes 16-31 hold rows m+8
  const int wm = wave & 3;
  const int wn = wave >> 2;
  const int mbase = blockIdx.y * 128 + wm * 32;
  const int nbase = blockIdx.x * 128 + wn * 64;

  const bf16_t* a0p = A + (size_t)(mbase + lr) * K;
  const bf16_t* a1p = A + (size_t)(mbase + 16 + lr) * K;
  const bf16_t* b0p = Bt + (size_t)(nbase +  0 + lr) * K + kbB;
  const bf16_t* b1p = Bt + (size_t)(nbase + 16 + lr) * K + kbB;
  const bf16_t* b2p = Bt + (size_t)(nbase + 32 + lr) * K + kbB;
  const bf16_t* b3p = Bt + (size_t)(nbase + 48 + lr) * K + kbB;

  v8f acc[2][4] = {};

  for (int k0 = 0; k0 < K; k0 += 32) {
    __builtin_prefetch(a0p + k0 + 512, 0, 0);   // global_prefetch_b8 (speculative)
    v16bf a0 = ld_a_frag(a0p + k0, kbA);
    v16bf a1 = ld_a_frag(a1p + k0, kbA);
    v16bf b0 = *(const v16bf*)(b0p + k0);
    v16bf b1 = *(const v16bf*)(b1p + k0);
    v16bf b2 = *(const v16bf*)(b2p + k0);
    v16bf b3 = *(const v16bf*)(b3p + k0);
    acc[0][0] = wmma_bf16(a0, b0, acc[0][0]);
    acc[0][1] = wmma_bf16(a0, b1, acc[0][1]);
    acc[0][2] = wmma_bf16(a0, b2, acc[0][2]);
    acc[0][3] = wmma_bf16(a0, b3, acc[0][3]);
    acc[1][0] = wmma_bf16(a1, b0, acc[1][0]);
    acc[1][1] = wmma_bf16(a1, b1, acc[1][1]);
    acc[1][2] = wmma_bf16(a1, b2, acc[1][2]);
    acc[1][3] = wmma_bf16(a1, b3, acc[1][3]);
  }

  #pragma unroll
  for (int mr = 0; mr < 2; ++mr)
    #pragma unroll
    for (int t = 0; t < 4; ++t)
      #pragma unroll
      for (int v = 0; v < 8; ++v) {
        int row = mbase + mr * 16 + v + rowoff;
        int col = nbase + t * 16 + lr;
        float val = acc[mr][t][v];
        if (mode == 0) {
          ((float*)out)[(size_t)row * N + col] = val;
        } else {
          int b = row >> 10, s = row & 1023;
          int h = col >> 6,  dk = col & 63;
          if (mode == 1)
            ((bf16_t*)out)[((size_t)(b * H_ + h) * S_ + s) * DK_ + dk] = (bf16_t)val;
          else
            ((bf16_t*)out)[((size_t)(b * H_ + h) * DK_ + dk) * S_ + s] = (bf16_t)val;
        }
      }
}

// ---------------------------------------------------------------------------
// Fused flash attention with T5 relative bias (no 1/sqrt(dk) scaling).
// grid = (S/64, B*H), block = 128 (4 independent waves, one 16-query tile each).
// Online softmax over 1024 keys in chunks of 32; scores never hit memory.
// ---------------------------------------------------------------------------
__global__ void __launch_bounds__(128)
k_flash(const bf16_t* __restrict__ Qh, const bf16_t* __restrict__ Kh,
        const bf16_t* __restrict__ Vt, const float* __restrict__ btab,
        bf16_t* __restrict__ ctx) {
  const int lane   = threadIdx.x & 31;
  const int wave   = threadIdx.x >> 5;
  const int lr     = lane & 15;
  const int kbA    = (lane < 16) ? 0 : 8;
  const int kbB    = (lane < 16) ? 0 : 16;
  const int rowoff = (lane < 16) ? 0 : 8;

  const int bh = blockIdx.y;          // b*H + h
  const int h  = bh & (H_ - 1);
  const int bb = bh >> 4;
  const int q0 = (blockIdx.x * 4 + wave) * 16;

  const bf16_t* Qp = Qh + (size_t)bh * S_ * DK_;
  const bf16_t* Kp = Kh + (size_t)bh * S_ * DK_;
  const bf16_t* Vp = Vt + (size_t)bh * DK_ * S_;

  // per-wave P-transpose staging tile, 16 rows x 32 cols bf16, stride 56
  // (112B rows: 16B-aligned for ds_load_b128, conflict-free across lane halves)
  __shared__ bf16_t pbuf[4][16 * 56] __attribute__((aligned(16)));
  bf16_t* myp = pbuf[wave];

  const bf16_t* qrow = Qp + (size_t)(q0 + lr) * DK_;
  v16bf aq0 = ld_a_frag(qrow, kbA);        // dk 0..31
  v16bf aq1 = ld_a_frag(qrow + 32, kbA);   // dk 32..63

  // bias(m,n) at key kc+t*16+lr, query q0+v+rowoff -> bp[kc + t*16 - v]
  const float* bp = btab + (h * 2047 + 1023 + lr - q0 - rowoff);

  float mrow[8], lrow[8];
  #pragma unroll
  for (int v = 0; v < 8; ++v) { mrow[v] = -1e30f; lrow[v] = 0.f; }
  v8f acc[4] = {};

  for (int kc = 0; kc < S_; kc += 32) {
    // ---- scores: two 16x16 tiles (keys kc..+15, kc+16..+31), K-dim = dk = 64
    const bf16_t* kr0 = Kp + (size_t)(kc + lr) * DK_ + kbB;
    const bf16_t* kr1 = Kp + (size_t)(kc + 16 + lr) * DK_ + kbB;
    v16bf bk00 = *(const v16bf*)(kr0);
    v16bf bk01 = *(const v16bf*)(kr0 + 32);
    v16bf bk10 = *(const v16bf*)(kr1);
    v16bf bk11 = *(const v16bf*)(kr1 + 32);
    v8f s0 = {}, s1 = {};
    s0 = wmma_bf16(aq0, bk00, s0);
    s0 = wmma_bf16(aq1, bk01, s0);
    s1 = wmma_bf16(aq0, bk10, s1);
    s1 = wmma_bf16(aq1, bk11, s1);

    // ---- bias + online softmax over this 32-key chunk
    float p0[8], p1[8], tm[8];
    #pragma unroll
    for (int v = 0; v < 8; ++v) {
      float x0 = s0[v] + bp[kc - v];
      float x1 = s1[v] + bp[kc + 16 - v];
      s0[v] = x0; s1[v] = x1;
      tm[v] = fmaxf(x0, x1);
    }
    #pragma unroll
    for (int v = 0; v < 8; ++v) {   // row-max across 16 lanes of each half
      tm[v] = fmaxf(tm[v], __shfl_xor(tm[v], 1, 16));
      tm[v] = fmaxf(tm[v], __shfl_xor(tm[v], 2, 16));
      tm[v] = fmaxf(tm[v], __shfl_xor(tm[v], 4, 16));
      tm[v] = fmaxf(tm[v], __shfl_xor(tm[v], 8, 16));
    }
    #pragma unroll
    for (int v = 0; v < 8; ++v) {
      float nm = fmaxf(mrow[v], tm[v]);
      float sc = __expf(mrow[v] - nm);
      p0[v] = __expf(s0[v] - nm);
      p1[v] = __expf(s1[v] - nm);
      float rs = p0[v] + p1[v];
      rs += __shfl_xor(rs, 1, 16);
      rs += __shfl_xor(rs, 2, 16);
      rs += __shfl_xor(rs, 4, 16);
      rs += __shfl_xor(rs, 8, 16);
      lrow[v] = lrow[v] * sc + rs;
      mrow[v] = nm;
      #pragma unroll
      for (int t = 0; t < 4; ++t) acc[t][v] *= sc;
    }

    // ---- P (C-layout) -> LDS row-major -> reload in A-layout (16x32 bf16)
    #pragma unroll
    for (int v = 0; v < 8; ++v) {
      myp[(v + rowoff) * 56 + lr]      = (bf16_t)p0[v];
      myp[(v + rowoff) * 56 + 16 + lr] = (bf16_t)p1[v];
    }
    asm volatile("s_wait_dscnt 0" ::: "memory");  // order store->load within wave
    v16bf pa = ld_a_frag(myp + lr * 56, kbA);

    // ---- O += P * V  (V^T rows are contiguous key slabs)
    #pragma unroll
    for (int t = 0; t < 4; ++t) {
      const bf16_t* vr = Vp + (size_t)(t * 16 + lr) * S_ + kc + kbB;
      v16bf bv = *(const v16bf*)vr;
      acc[t] = wmma_bf16(pa, bv, acc[t]);
    }
  }

  // ---- normalize and store ctx as bf16 [b, s, h*64+dk] (A operand of out GEMM)
  #pragma unroll
  for (int v = 0; v < 8; ++v) lrow[v] = 1.0f / lrow[v];
  #pragma unroll
  for (int t = 0; t < 4; ++t)
    #pragma unroll
    for (int v = 0; v < 8; ++v) {
      int srow = q0 + v + rowoff;
      ctx[((size_t)(bb * S_ + srow)) * (H_ * DK_) + h * DK_ + t * 16 + lr] =
          (bf16_t)(acc[t][v] * lrow[v]);
    }
}

// ---------------------------------------------------------------------------
// Launch
// ---------------------------------------------------------------------------
extern "C" void kernel_launch(void* const* d_in, const int* in_sizes, int n_in,
                              void* d_out, int out_size, void* d_ws, size_t ws_size,
                              hipStream_t stream) {
  const float* hidden   = (const float*)d_in[0];
  const float* wq       = (const float*)d_in[1];
  const float* wk       = (const float*)d_in[2];
  const float* wv       = (const float*)d_in[3];
  const float* wo       = (const float*)d_in[4];
  const float* rel_bias = (const float*)d_in[5];

  char* ws = (char*)d_ws;
  const size_t MB = 1024 * 1024;
  bf16_t* xb   = (bf16_t*)(ws + 0 * MB);    // X bf16            16 MB
  bf16_t* wqt  = (bf16_t*)(ws + 16 * MB);   // wq^T bf16          2 MB
  bf16_t* wkt  = (bf16_t*)(ws + 18 * MB);
  bf16_t* wvt  = (bf16_t*)(ws + 20 * MB);
  bf16_t* wot  = (bf16_t*)(ws + 22 * MB);
  bf16_t* qh   = (bf16_t*)(ws + 24 * MB);   // Q [b,h,s,dk]      16 MB
  bf16_t* kh   = (bf16_t*)(ws + 40 * MB);   // K [b,h,s,dk]      16 MB
  bf16_t* vt   = (bf16_t*)(ws + 56 * MB);   // V [b,h,dk,s]      16 MB
  bf16_t* ctxb = (bf16_t*)(ws + 72 * MB);   // ctx [b*s, h*dk]   16 MB
  float*  tab  = (float*)(ws + 88 * MB);    // bias table        128 KB

  // prep
  k_cvt_bf16<<<1024, 256, 0, stream>>>(hidden, xb, B_ * S_ * D_);
  k_transpose_w<<<4096, 256, 0, stream>>>(wq, wqt);
  k_transpose_w<<<4096, 256, 0, stream>>>(wk, wkt);
  k_transpose_w<<<4096, 256, 0, stream>>>(wv, wvt);
  k_transpose_w<<<4096, 256, 0, stream>>>(wo, wot);
  k_bias_tab<<<(H_ * 2047 + 255) / 256, 256, 0, stream>>>(rel_bias, tab);

  // QKV projections (M=8192, N=1024, K=1024)
  dim3 gg(1024 / 128, (B_ * S_) / 128);
  k_gemm_bf16<<<gg, 256, 0, stream>>>(xb, wqt, qh, B_ * S_, 1024, 1024, 1);
  k_gemm_bf16<<<gg, 256, 0, stream>>>(xb, wkt, kh, B_ * S_, 1024, 1024, 1);
  k_gemm_bf16<<<gg, 256, 0, stream>>>(xb, wvt, vt, B_ * S_, 1024, 1024, 2);

  // fused attention
  dim3 gf(S_ / 64, B_ * H_);
  k_flash<<<gf, 128, 0, stream>>>(qh, kh, vt, tab, ctxb);

  // output projection -> fp32
  k_gemm_bf16<<<gg, 256, 0, stream>>>(ctxb, wot, d_out, B_ * S_, 1024, 1024, 0);
}
